// ConvDiscriminator_53712861004596
// MI455X (gfx1250) — compile-verified, matrix operand order checked by reading, not run on hardware
//
#include <hip/hip_runtime.h>
#include <math.h>

// CDNA5 (gfx1250) wave32 WMMA types
typedef __attribute__((ext_vector_type(16))) __bf16 v16bf;
typedef __attribute__((ext_vector_type(8)))  float  v8f;

// Optional gfx1250 async global->LDS path (guarded; falls back to plain copy)
#if defined(__has_builtin)
#if __has_builtin(__builtin_amdgcn_global_load_async_to_lds_b128)
#define USE_ASYNC_LDS 1
#endif
#endif
#ifndef USE_ASYNC_LDS
#define USE_ASYNC_LDS 0
#endif

#if USE_ASYNC_LDS
typedef int v4i __attribute__((vector_size(16)));
typedef __attribute__((address_space(1))) v4i* gv4p;  // global
typedef __attribute__((address_space(3))) v4i* sv4p;  // LDS
#endif

// ---------- helpers ----------
__device__ __forceinline__ unsigned short f2bf(float f) {
  unsigned u = __float_as_uint(f);
  unsigned r = (u + 0x7FFFu + ((u >> 16) & 1u)) >> 16;  // RNE
  return (unsigned short)r;
}

// A-tile element offset for a 16xK activation matrix stored as WMMA A tiles.
// Lane l holds M=l&15, K-half=(l>>4); each lane owns 16 contiguous bf16 (32B).
__device__ __forceinline__ int a_off(int m, int k) {
  return ((k >> 5) * 32 + ((k >> 4) & 1) * 16 + m) * 16 + (k & 15);
}

// ---------- weight packing: fp32 [K][N] row-major -> bf16 B-tiles ----------
__global__ void pack_b_kernel(const float* __restrict__ W,
                              unsigned short* __restrict__ Bt, int K, int N) {
  int idx = blockIdx.x * 256 + threadIdx.x;
  if (idx >= K * N) return;
  int j = idx & 15;
  int l = (idx >> 4) & 31;
  int rest = idx >> 9;
  int KT = K >> 5;
  int kt = rest % KT;
  int nt = rest / KT;
  int n = nt * 16 + (l & 15);
  int k = kt * 32 + ((l >> 4) << 4) + j;
  Bt[idx] = f2bf(W[k * N + n]);
}

// x0 = embedding[SOS=1] broadcast to 16 rows, packed as A-tiles (16x512 bf16)
__global__ void pack_x0a_kernel(const float* __restrict__ emb,
                                unsigned short* __restrict__ x0A) {
  int idx = blockIdx.x * 256 + threadIdx.x;
  if (idx >= 16 * 512) return;
  int m = idx >> 9, k = idx & 511;
  x0A[a_off(m, k)] = f2bf(emb[1 * 512 + k]);
}

// ---------- encoder: enc_out[l][b][h] = tanh(emb[input[b,l]] @ enc_W + b) ----------
__global__ void enc_kernel(const int* __restrict__ inp, const float* __restrict__ emb,
                           const unsigned short* __restrict__ Wt,
                           const float* __restrict__ bias, float* __restrict__ enc_out) {
  __shared__ unsigned short Ash[16 * 512];  // 16KB
  __shared__ int tok[16];
  int tid = threadIdx.x;
  int mtile = blockIdx.x;
  if (tid < 16) tok[tid] = inp[mtile * 16 + tid];
  __syncthreads();
  for (int e = tid; e < 16 * 512; e += 256) {
    int m = e >> 9, k = e & 511;
    Ash[a_off(m, k)] = f2bf(emb[(size_t)tok[m] * 512 + k]);
  }
  __syncthreads();
  int wave = tid >> 5, lane = tid & 31;
  int nt = blockIdx.y * 8 + wave;
  v8f acc = {};
  const v16bf* Arow = (const v16bf*)Ash;
  const unsigned short* bp = Wt + ((size_t)nt * 16) * 512 + lane * 16;
  for (int kt = 0; kt < 16; ++kt) {  // K=512
    v16bf a = Arow[kt * 32 + lane];
    v16bf b = *(const v16bf*)(bp + kt * 512);
    acc = __builtin_amdgcn_wmma_f32_16x16x32_bf16(false, a, false, b,
                                                  (short)0, acc, false, false);
  }
  int n = nt * 16 + (lane & 15);
  float bn = bias[n];
  for (int r = 0; r < 8; ++r) {
    int m = r + (lane >> 4) * 8;
    int tokidx = mtile * 16 + m;
    int bb = tokidx >> 7, ll = tokidx & 127;
    enc_out[((size_t)(ll * 16 + bb)) * 1024 + n] = tanhf(acc[r] + bn);
  }
}

// ---------- h_init: h0 = h1 = enc_out[L-1] (+ bf16 A-tiles) ----------
__global__ void hinit_kernel(const float* __restrict__ enc_out, float* __restrict__ h0,
                             float* __restrict__ h1, unsigned short* __restrict__ h0A,
                             unsigned short* __restrict__ h1A) {
  int idx = blockIdx.x * 256 + threadIdx.x;  // 16384
  int b = idx >> 10, n = idx & 1023;
  float v = enc_out[((size_t)(127 * 16 + b)) * 1024 + n];
  h0[idx] = v;
  h1[idx] = v;
  unsigned short bf = f2bf(v);
  h0A[a_off(b, n)] = bf;
  h1A[a_off(b, n)] = bf;
}

// ---------- generic M=16 GEMM body: C[16][N] = A(16xK bf16 tiles) @ B(tiles) ----------
__device__ __forceinline__ void gemm16_body(const unsigned short* __restrict__ A,
                                            const unsigned short* __restrict__ Bt,
                                            float* __restrict__ C, int K, int N,
                                            unsigned short* Ash, int ntile) {
  int tid = threadIdx.x;
  int nvec = K * 2;  // (16*K*2 bytes)/16
#if USE_ASYNC_LDS
  {
    const uint4* src = (const uint4*)A;
    uint4* dst = (uint4*)Ash;
    for (int i = tid; i < nvec; i += 256) {
      v4i* sp = (v4i*)(src + i);  // strips const, still flat
      v4i* dp = (v4i*)(dst + i);
      __builtin_amdgcn_global_load_async_to_lds_b128((gv4p)sp, (sv4p)dp, 0, 0);
    }
#if defined(__has_builtin) && __has_builtin(__builtin_amdgcn_s_wait_asynccnt)
    __builtin_amdgcn_s_wait_asynccnt(0);
#else
    asm volatile("s_wait_asynccnt 0" ::: "memory");
#endif
  }
#else
  {
    const uint4* src = (const uint4*)A;
    uint4* dst = (uint4*)Ash;
    for (int i = tid; i < nvec; i += 256) dst[i] = src[i];
  }
#endif
  __syncthreads();
  int wave = tid >> 5, lane = tid & 31;
  int nt = ntile * 8 + wave;
  int KT = K >> 5;
  v8f acc = {};
  const v16bf* Arow = (const v16bf*)Ash;
  const unsigned short* bp = Bt + (size_t)nt * KT * 512 + lane * 16;
  for (int kt = 0; kt < KT; ++kt) {
    if (kt + 1 < KT) __builtin_prefetch(bp + (kt + 1) * 512, 0, 3);  // global_prefetch_b8
    v16bf a = Arow[kt * 32 + lane];
    v16bf b = *(const v16bf*)(bp + kt * 512);
    acc = __builtin_amdgcn_wmma_f32_16x16x32_bf16(false, a, false, b,
                                                  (short)0, acc, false, false);
  }
  int n = nt * 16 + (lane & 15);
  for (int r = 0; r < 8; ++r) {
    int m = r + (lane >> 4) * 8;
    C[(size_t)m * N + n] = acc[r];
  }
}

__global__ void gemm16_kernel(const unsigned short* __restrict__ A,
                              const unsigned short* __restrict__ Bt,
                              float* __restrict__ C, int K, int N) {
  __shared__ unsigned short Ash[16 * 1024];  // up to 32KB
  gemm16_body(A, Bt, C, K, N, Ash, blockIdx.x);
}

// Two independent M=16 GEMMs in one launch (gridDim.y == 2)
__global__ void gemm16_dual_kernel(const unsigned short* __restrict__ A0,
                                   const unsigned short* __restrict__ B0,
                                   float* __restrict__ C0,
                                   const unsigned short* __restrict__ A1,
                                   const unsigned short* __restrict__ B1,
                                   float* __restrict__ C1, int K, int N) {
  __shared__ unsigned short Ash[16 * 1024];
  if (blockIdx.y == 0)
    gemm16_body(A0, B0, C0, K, N, Ash, blockIdx.x);
  else
    gemm16_body(A1, B1, C1, K, N, Ash, blockIdx.x);
}

// ---------- GRU gate combine: h = (1-z)*tanh(xn + r*hn) + z*h ----------
__global__ void gru_combine_kernel(const float* __restrict__ gx, const float* __restrict__ bih,
                                   const float* __restrict__ gh, const float* __restrict__ bhh,
                                   const float* __restrict__ hin, float* __restrict__ hout,
                                   unsigned short* __restrict__ hA) {
  int idx = blockIdx.x * 256 + threadIdx.x;  // 16384
  int m = idx >> 10, n = idx & 1023;
  int base = m * 3072;
  float xr = gx[base + n] + bih[n];
  float xz = gx[base + 1024 + n] + bih[1024 + n];
  float xn = gx[base + 2048 + n] + bih[2048 + n];
  float hr = gh[base + n] + bhh[n];
  float hz = gh[base + 1024 + n] + bhh[1024 + n];
  float hn = gh[base + 2048 + n] + bhh[2048 + n];
  float r = 1.f / (1.f + __expf(-(xr + hr)));
  float z = 1.f / (1.f + __expf(-(xz + hz)));
  float nn = tanhf(xn + r * hn);
  float h = (1.f - z) * nn + z * hin[m * 1024 + n];
  hout[m * 1024 + n] = h;
  hA[a_off(m, n)] = f2bf(h);
}

// ---------- attention + context + gathered output, one block per batch ----------
__global__ void attn_out_kernel(const float* __restrict__ enc, const float* __restrict__ h1,
                                const int* __restrict__ inp, const float* __restrict__ Wout,
                                const float* __restrict__ bout, float* __restrict__ out, int t) {
  __shared__ float hsh[1024];
  __shared__ float sc[128];
  __shared__ float ctx[1024];
  __shared__ float red[256];
  int b = blockIdx.x, tid = threadIdx.x;
  for (int i = tid; i < 1024; i += 256) hsh[i] = h1[b * 1024 + i];
  __syncthreads();
  int wave = tid >> 5, lane = tid & 31;
  for (int l = wave; l < 128; l += 8) {
    const float* er = enc + ((size_t)(l * 16 + b)) * 1024;
    float s = 0.f;
    for (int h = lane; h < 1024; h += 32) s += hsh[h] * er[h];
    for (int o = 16; o > 0; o >>= 1) s += __shfl_xor(s, o, 32);
    if (lane == 0) sc[l] = s * 0.03125f;  // 1/sqrt(1024)
  }
  __syncthreads();
  float mx = -1e30f;
  for (int l = 0; l < 128; ++l) mx = fmaxf(mx, sc[l]);
  float sum = 0.f;
  for (int l = 0; l < 128; ++l) sum += __expf(sc[l] - mx);
  __syncthreads();
  if (tid < 128) sc[tid] = __expf(sc[tid] - mx) / sum;
  __syncthreads();
  for (int h = tid; h < 1024; h += 256) {
    float c = 0.f;
    for (int l = 0; l < 128; ++l) c += sc[l] * enc[((size_t)(l * 16 + b)) * 1024 + h];
    ctx[h] = c;
  }
  __syncthreads();
  int col = inp[b * 128 + t];
  float d = 0.f;
  for (int h = tid; h < 1024; h += 256) {
    d += hsh[h] * Wout[(size_t)h * 32000 + col];
    d += ctx[h] * Wout[(size_t)(1024 + h) * 32000 + col];
  }
  red[tid] = d;
  __syncthreads();
  for (int o = 128; o > 0; o >>= 1) {
    if (tid < o) red[tid] += red[tid + o];
    __syncthreads();
  }
  if (tid == 0) {
    float logit = red[0] + bout[col];
    out[1 + b * 128 + t] = 1.f / (1.f + __expf(-logit));  // sigmoid prob
  }
}

// ---------- final BCE loss over the 2048 gathered probs ----------
__global__ void loss_kernel(float* __restrict__ out, const int* __restrict__ ts) {
  __shared__ float red[256];
  int tid = threadIdx.x;
  float tv = (float)(*ts);
  float s = 0.f;
  for (int i = tid; i < 2048; i += 256) {
    float p = out[1 + i];
    p = fminf(fmaxf(p, 1e-12f), 1.0f - 1e-7f);
    s += -(tv * logf(p) + (1.f - tv) * log1pf(-p));
  }
  red[tid] = s;
  __syncthreads();
  for (int o = 128; o > 0; o >>= 1) {
    if (tid < o) red[tid] += red[tid + o];
    __syncthreads();
  }
  if (tid == 0) out[0] = red[0] / 2048.f / 16.f;  // mean(bce)/batch_size
}

extern "C" void kernel_launch(void* const* d_in, const int* in_sizes, int n_in,
                              void* d_out, int out_size, void* d_ws, size_t ws_size,
                              hipStream_t stream) {
  const int*   input = (const int*)d_in[0];
  const int*   tsamp = (const int*)d_in[2];
  const float* emb   = (const float*)d_in[3];
  const float* encW  = (const float*)d_in[4];
  const float* encb  = (const float*)d_in[5];
  const float* Wih0  = (const float*)d_in[6];
  const float* Whh0  = (const float*)d_in[7];
  const float* bih0  = (const float*)d_in[8];
  const float* bhh0  = (const float*)d_in[9];
  const float* Wih1  = (const float*)d_in[10];
  const float* Whh1  = (const float*)d_in[11];
  const float* bih1  = (const float*)d_in[12];
  const float* bhh1  = (const float*)d_in[13];
  const float* Wout  = (const float*)d_in[14];
  const float* bout  = (const float*)d_in[15];
  float* out = (float*)d_out;
  char* ws = (char*)d_ws;

  // workspace layout (bytes, 256B aligned)
  unsigned short* Whh0_t = (unsigned short*)(ws + 0);         // 1024x3072 bf16
  unsigned short* Wih1_t = (unsigned short*)(ws + 6291456);
  unsigned short* Whh1_t = (unsigned short*)(ws + 12582912);
  unsigned short* encW_t = (unsigned short*)(ws + 18874368);  // 512x1024
  unsigned short* Wih0_t = (unsigned short*)(ws + 19922944);  // 512x3072
  float* enc_out = (float*)(ws + 23068672);                   // 128x16x1024 f32
  float* gx0     = (float*)(ws + 31457280);                   // 16x3072
  float* gh0     = (float*)(ws + 31653888);
  float* gx1     = (float*)(ws + 31850496);
  float* gh1     = (float*)(ws + 32047104);
  float* h0      = (float*)(ws + 32243712);                   // 16x1024
  float* h1      = (float*)(ws + 32309248);
  unsigned short* h0A = (unsigned short*)(ws + 32374784);     // 16x1024 bf16
  unsigned short* h1A = (unsigned short*)(ws + 32407552);
  unsigned short* x0A = (unsigned short*)(ws + 32440320);     // 16x512 bf16

  // ---- one-time setup ----
  pack_b_kernel<<<(1024 * 3072) / 256, 256, 0, stream>>>(Whh0, Whh0_t, 1024, 3072);
  pack_b_kernel<<<(1024 * 3072) / 256, 256, 0, stream>>>(Wih1, Wih1_t, 1024, 3072);
  pack_b_kernel<<<(1024 * 3072) / 256, 256, 0, stream>>>(Whh1, Whh1_t, 1024, 3072);
  pack_b_kernel<<<(512 * 1024) / 256, 256, 0, stream>>>(encW, encW_t, 512, 1024);
  pack_b_kernel<<<(512 * 3072) / 256, 256, 0, stream>>>(Wih0, Wih0_t, 512, 3072);
  pack_x0a_kernel<<<32, 256, 0, stream>>>(emb, x0A);
  enc_kernel<<<dim3(128, 8), 256, 0, stream>>>(input, emb, encW_t, encb, enc_out);
  // gx0 = x0 @ Wih0 (loop-invariant: decoder input is constant SOS)
  gemm16_kernel<<<24, 256, 0, stream>>>(x0A, Wih0_t, gx0, 512, 3072);
  hinit_kernel<<<64, 256, 0, stream>>>(enc_out, h0, h1, h0A, h1A);

  // ---- 128 decode steps ----
  for (int t = 0; t < 128; ++t) {
    gemm16_kernel<<<24, 256, 0, stream>>>(h0A, Whh0_t, gh0, 1024, 3072);
    gru_combine_kernel<<<64, 256, 0, stream>>>(gx0, bih0, gh0, bhh0, h0, h0, h0A);
    gemm16_dual_kernel<<<dim3(24, 2), 256, 0, stream>>>(h0A, Wih1_t, gx1,
                                                        h1A, Whh1_t, gh1, 1024, 3072);
    gru_combine_kernel<<<64, 256, 0, stream>>>(gx1, bih1, gh1, bhh1, h1, h1, h1A);
    attn_out_kernel<<<16, 256, 0, stream>>>(enc_out, h1, input, Wout, bout, out, t);
  }
  loss_kernel<<<1, 256, 0, stream>>>(out, tsamp);
}
